// Downsampling_16930761081224
// MI455X (gfx1250) — matrix operand order repaired and checked by reading, not compiled, not run on hardware
//
#include <hip/hip_runtime.h>
#include <hip/hip_bf16.h>

typedef _Float16 h16;
typedef __attribute__((ext_vector_type(16))) _Float16 v16h;
typedef __attribute__((ext_vector_type(8)))  _Float16 v8h;
typedef __attribute__((ext_vector_type(8)))  float    v8f;

#define DEV static __device__ __forceinline__

// ---------------------------------------------------------------------------
// WMMA helpers (CDNA5 wave32, V_WMMA_F32_16X16X32_F16)
// A fragment (16x32 f16): lanes 0-15 -> M=lane, K in {0..7,16..23};
//                         lanes 16-31 -> M=lane-16, K in {8..15,24..31}
// B fragment (32x16 f16): lanes 0-15 -> N=lane, K in {0..7,16..23}; lanes 16-31 -> K in {8..15,24..31}
// D (16x16 f32, 8 VGPR): lanes 0-15 -> N=lane, M=i; lanes 16-31 -> M=8+i
// ---------------------------------------------------------------------------
DEV v8f wmma16(v16h a, v16h b, v8f c) {
    return __builtin_amdgcn_wmma_f32_16x16x32_f16(false, a, false, b, (short)0, c,
                                                  false, false);
}

DEV v16h cat8(v8h lo, v8h hi) {
    return __builtin_shufflevector(lo, hi, 0, 1, 2, 3, 4, 5, 6, 7,
                                   8, 9, 10, 11, 12, 13, 14, 15);
}

// A tile in LDS, row-major [m][k], ld = 32 halfs.  Two aligned v8h loads.
DEV v16h frag_a_lds(const h16* base) {
    const int lane = threadIdx.x & 31;
    const int m  = lane & 15;
    const int kb = (lane & 16) ? 16 : 0;
    const h16* p = base + m * 32 + kb;
    return cat8(*(const v8h*)p, *(const v8h*)(p + 8));
}

// B tile in LDS, TRANSPOSED [n][k], ld = 32 halfs.  Two aligned v8h loads.
DEV v16h frag_bt_lds(const h16* Bt, int nsub /* 16-col group */) {
    const int lane = threadIdx.x & 31;
    const int n  = lane & 15;
    const int kb = (lane & 16) ? 16 : 0;
    const h16* p = Bt + (nsub * 16 + n) * 32 + kb;
    return cat8(*(const v8h*)p, *(const v8h*)(p + 8));
}

// 16 x 160 weight in LDS as [n][k].  Two aligned v8h loads.
DEV v16h frag_bt160(const h16* Wt, int kc) {
    const int lane = threadIdx.x & 31;
    const int n  = lane & 15;
    const int kb = kc * 32 + ((lane & 16) ? 8 : 0);
    const h16* p = Wt + n * 160 + kb;
    return cat8(*(const v8h*)p, *(const v8h*)(p + 16));
}

DEV float fastrcp(float x)   { return __builtin_amdgcn_rcpf(x); }
DEV float sigmoidf_(float x) { return fastrcp(1.f + __expf(-x)); }
DEV float siluf_(float x)    { return x * sigmoidf_(x); }

DEV unsigned lds_off(const void* p) { return (unsigned)(unsigned long long)p; }

// Async global->LDS copy of 8 bytes per lane (ASYNCcnt tracked).
DEV void async_b64(unsigned lds_addr, const void* gaddr) {
    asm volatile("global_load_async_to_lds_b64 %0, %1, off"
                 :: "v"(lds_addr), "v"(gaddr) : "memory");
}
DEV void wait_async0() {
    asm volatile("s_wait_asynccnt 0" ::: "memory");
}

// ---------------------------------------------------------------------------
// Templated tiled WMMA GEMM: C[M,N] = A[M,K] @ B[K,N] + bias
// B is f16 stored TRANSPOSED (N-major): BwT[n*K + k]  -> async b64 staging.
// A-fetch MODE:  0 plain f32 row-major (lda=K)
//                1 concat: [content(256) ; style(128)], rows = b*512+s
//                2 im2col3: K = tap*256 + c  -> V[b, s+tap-1, c] (zero pad)
// OUTF: f32 output C[grow*N+gcol]; OUTH: f16 transposed per-batch output
//       ChT[((grow>>9)*N + gcol)*512 + (grow&511)]   (for sP staging reuse)
// Block: 256 thr = 8 waves; tile 64(M) x 64(N); K chunks of 32; 2 WMMA/wave.
// ---------------------------------------------------------------------------
template<int MODE, bool BIAS, bool ALPHA, bool OUTF, bool OUTH>
__global__ __launch_bounds__(256)
void gemm_wmma(const float* __restrict__ A, const float* __restrict__ style,
               const h16* __restrict__ BwT, const float* __restrict__ bias,
               float* __restrict__ C, h16* __restrict__ ChT,
               const float* __restrict__ alpha,
               int M, int N, int K,
               long long sA, long long sB, long long sC)
{
    const int bz = blockIdx.z;
    const float* Ab = A + (long long)bz * sA;
    const h16*   Bb = BwT + (long long)bz * sB;

    __shared__ __align__(16) h16 As[64 * 32];
    __shared__ __align__(16) h16 Bt[64 * 32];   // transposed [n][k]

    const int tid  = threadIdx.x;
    const int wave = tid >> 5;
    const int row0 = blockIdx.y * 64;
    const int col0 = blockIdx.x * 64;
    const int wm = wave >> 1, wn = wave & 1;
    const unsigned ldsB = lds_off(&Bt[0]);

    v8f acc0 = {}, acc1 = {};

    for (int k0 = 0; k0 < K; k0 += 32) {
        // ---- stage B (64 rows x 32 k) via async global->LDS b64 copies ----
        {
#pragma unroll
            for (int u = 0; u < 2; ++u) {
                const int cid = tid * 2 + u;            // 0..511 b64 chunks
                const int n = cid >> 3, k8 = (cid & 7) * 4;
                const h16* gp = Bb + (long long)(col0 + n) * K + k0 + k8;
                async_b64(ldsB + (unsigned)((n * 32 + k8) * 2), gp);
            }
        }
        // ---- stage A (64x32): each thread converts 8 contiguous elements ----
        {
            const int i = tid * 8;
            const int r = i >> 5, kk = i & 31;
            const int grow = row0 + r;
            const int gk0 = k0 + kk;
            float v[8];
            if (MODE == 0) {
                const float* p = &Ab[(long long)grow * K + gk0];
#pragma unroll
                for (int u = 0; u < 8; ++u) v[u] = p[u];
                if (k0 + 32 < K) __builtin_prefetch(p + 32, 0, 3);
            } else if (MODE == 1) {
                if (gk0 < 256) {
                    const float* p = &Ab[(long long)grow * 256 + gk0];
#pragma unroll
                    for (int u = 0; u < 8; ++u) v[u] = p[u];
                } else {
                    const float* p = &style[(grow >> 9) * 128 + (gk0 - 256)];
#pragma unroll
                    for (int u = 0; u < 8; ++u) v[u] = p[u];
                }
            } else {
                const int b = grow >> 9, s = grow & 511;
                const int tap = gk0 >> 8, c0 = gk0 & 255;   // group-uniform
                const int ss = s + tap - 1;
                if (ss < 0 || ss >= 512) {
#pragma unroll
                    for (int u = 0; u < 8; ++u) v[u] = 0.f;
                } else {
                    const float* p = &Ab[((long long)(b << 9) + ss) * 256 + c0];
#pragma unroll
                    for (int u = 0; u < 8; ++u) v[u] = p[u];
                    if (k0 + 32 < K) __builtin_prefetch(p + 256, 0, 3);
                }
            }
            v8h hv;
#pragma unroll
            for (int u = 0; u < 8; ++u) hv[u] = (h16)v[u];
            *(v8h*)&As[i] = hv;
        }
        wait_async0();
        __syncthreads();
        v16h af = frag_a_lds(&As[wm * 16 * 32]);
        v16h b0 = frag_bt_lds(Bt, wn * 2 + 0);
        v16h b1 = frag_bt_lds(Bt, wn * 2 + 1);
        acc0 = wmma16(af, b0, acc0);
        acc1 = wmma16(af, b1, acc1);
        __syncthreads();
    }

    const int lane = tid & 31;
    const int n  = lane & 15;
    const int mb = (lane & 16) ? 8 : 0;
    const int gcol0 = col0 + (wn * 2) * 16 + n;
    const int gcol1 = gcol0 + 16;
    const float bv0 = BIAS ? bias[gcol0] : 0.f;
    const float bv1 = BIAS ? bias[gcol1] : 0.f;
#pragma unroll
    for (int i = 0; i < 8; ++i) {
        const int grow = row0 + wm * 16 + mb + i;
        float v0 = acc0[i] + bv0;
        float v1 = acc1[i] + bv1;
        if (ALPHA) { float al = alpha[grow >> 9]; v0 *= al; v1 *= al; }
        if (OUTF) {
            const long long base = (long long)grow * N + (long long)bz * sC;
            C[base + gcol0] = v0;
            C[base + gcol1] = v1;
        }
        if (OUTH) {   // transposed per-batch f16: [(b*N + n)*512 + s]
            const long long bb = (long long)(grow >> 9) * N;
            const int s = grow & 511;
            ChT[(bb + gcol0) * 512 + s] = (h16)v0;
            ChT[(bb + gcol1) * 512 + s] = (h16)v1;
        }
    }
}

// ---------------------------------------------------------------------------
// Weight conversion / repacking (all B operands stored N-major: dst[n*K + k])
// ---------------------------------------------------------------------------
__global__ void k_cvt_t(const float* __restrict__ src, h16* __restrict__ dst,
                        int K, int N) {
    int i = blockIdx.x * blockDim.x + threadIdx.x;  // i = n*K + k
    if (i >= K * N) return;
    int n = i / K, k = i - n * K;
    dst[i] = (h16)src[k * N + n];
}

// p2_w (256,256,3) -> dst[n*768 + tap*256 + cin]
__global__ void k_pack_p2(const float* __restrict__ w, h16* __restrict__ d) {
    int i = blockIdx.x * blockDim.x + threadIdx.x;
    if (i >= 256 * 768) return;
    int n = i / 768, k = i - n * 768;
    int tap = k >> 8, cin = k & 255;
    d[i] = (h16)w[((n << 8) + cin) * 3 + tap];
}

// mkb_w (16, cin, 3, 3) -> dst[n*160 + tap*16 + c], K padded 144->160
__global__ void k_pack_mkb(const float* __restrict__ w, h16* __restrict__ d, int cin) {
    int i = blockIdx.x * blockDim.x + threadIdx.x;
    if (i >= 16 * 160) return;
    int n = i / 160, k = i - n * 160;
    int tap = k >> 4, c = k & 15;
    float v = 0.f;
    if (tap < 9 && c < cin) {
        int dt = tap / 3, ds = tap - dt * 3;
        v = w[((n * cin + c) * 3 + dt) * 3 + ds];
    }
    d[i] = (h16)v;
}

// ---------------------------------------------------------------------------
// Elementwise / row kernels
// ---------------------------------------------------------------------------
__global__ __launch_bounds__(256)
void k_glu_mask(const float* __restrict__ h512, const unsigned char* __restrict__ mask,
                float* __restrict__ out) {
    long long i = (long long)blockIdx.x * 256 + threadIdx.x;
    if (i >= (long long)8192 * 256) return;
    int row = (int)(i >> 8), c = (int)(i & 255);
    float a = h512[(long long)row * 512 + c];
    float g = h512[(long long)row * 512 + 256 + c];
    out[i] = mask[row] ? 0.f : a * sigmoidf_(g);
}

__global__ __launch_bounds__(256)
void k_dwconv(const float* __restrict__ hc, const float* __restrict__ lck,
              float* __restrict__ out) {
    long long i = (long long)blockIdx.x * 256 + threadIdx.x;
    if (i >= (long long)8192 * 256) return;
    int row = (int)(i >> 8), c = (int)(i & 255);
    int s = row & 511;
    int head = c >> 5;
    float k0 = lck[head * 3 + 0], k1 = lck[head * 3 + 1], k2 = lck[head * 3 + 2];
    float mx = fmaxf(k0, fmaxf(k1, k2));
    float e0 = __expf(k0 - mx), e1 = __expf(k1 - mx), e2 = __expf(k2 - mx);
    float inv = fastrcp(e0 + e1 + e2);
    float xm = (s > 0)   ? hc[i - 256] : 0.f;
    float x0 = hc[i];
    float xp = (s < 511) ? hc[i + 256] : 0.f;
    out[i] = (e0 * xm + e1 * x0 + e2 * xp) * inv;
}

__global__ __launch_bounds__(256)
void k_add_ln_mask(const float* __restrict__ res, const float* __restrict__ h,
                   const float* __restrict__ g, const float* __restrict__ bb,
                   const unsigned char* __restrict__ mask, float* __restrict__ out) {
    int row = blockIdx.x;
    int c = threadIdx.x;
    __shared__ float red[256];
    long long idx = (long long)row * 256 + c;
    float v = res[idx] + h[idx];
    red[c] = v; __syncthreads();
    for (int st = 128; st > 0; st >>= 1) { if (c < st) red[c] += red[c + st]; __syncthreads(); }
    float mean = red[0] * (1.f / 256.f); __syncthreads();
    float d = v - mean;
    red[c] = d * d; __syncthreads();
    for (int st = 128; st > 0; st >>= 1) { if (c < st) red[c] += red[c + st]; __syncthreads(); }
    float var = red[0] * (1.f / 256.f);
    float y = d * rsqrtf(var + 1e-5f) * g[c] + bb[c];
    out[idx] = mask[row] ? 0.f : y;
}

__global__ __launch_bounds__(256)
void k_ce_ln(const float* __restrict__ cer, const float* __restrict__ g,
             const float* __restrict__ bb, const int* __restrict__ tlen,
             float* __restrict__ out) {
    int row = blockIdx.x;                // b*256 + t
    int b = row >> 8, t = row & 255;
    int c = threadIdx.x;
    __shared__ float red[256];
    long long idx = (long long)row * 256 + c;
    float v = cer[idx];
    red[c] = v; __syncthreads();
    for (int st = 128; st > 0; st >>= 1) { if (c < st) red[c] += red[c + st]; __syncthreads(); }
    float mean = red[0] * (1.f / 256.f); __syncthreads();
    float d = v - mean;
    red[c] = d * d; __syncthreads();
    for (int st = 128; st > 0; st >>= 1) { if (c < st) red[c] += red[c + st]; __syncthreads(); }
    float var = red[0] * (1.f / 256.f);
    float y = d * rsqrtf(var + 1e-5f) * g[c] + bb[c];
    out[idx] = (t >= tlen[b]) ? 0.f : y;
}

// ---------------------------------------------------------------------------
// Duration head: D = softplus(V @ p1_w + b), masked; alpha; sD; cumsum; text_len
// ---------------------------------------------------------------------------
__global__ __launch_bounds__(512)
void k_duration(const float* __restrict__ V, const float* __restrict__ p1w,
                const float* __restrict__ p1b, const unsigned char* __restrict__ mask,
                const float* __restrict__ rA, const float* __restrict__ mel,
                float* __restrict__ out_alpha, float* __restrict__ out_sD,
                float* __restrict__ skb, float* __restrict__ ekb,
                int* __restrict__ tlen, float* __restrict__ alw)
{
    int b = blockIdx.x, s = threadIdx.x;
    __shared__ float sh[512];
    float d = 0.f;
    if (!mask[b * 512 + s]) {
        float a = 0.f;
        const float4* vp = (const float4*)(V + ((long long)b * 512 + s) * 256);
        const float4* wp = (const float4*)p1w;
        for (int k = 0; k < 64; ++k) {
            float4 x = vp[k], w = wp[k];
            a += x.x * w.x + x.y * w.y + x.z * w.z + x.w * w.w;
        }
        a += p1b[0];
        d = (a > 20.f) ? a : log1pf(__expf(a));     // softplus
    }
    sh[s] = d; __syncthreads();
    for (int off = 1; off < 512; off <<= 1) {       // inclusive scan
        float tv = (s >= off) ? sh[s - off] : 0.f;
        __syncthreads();
        sh[s] += tv;
        __syncthreads();
    }
    float totalD = sh[511];
    float alpha = mel[b] * rA[b] / totalD;
    float sD = alpha * d;
    float ek = alpha * sh[s];
    out_sD[b * 512 + s] = sD;
    ekb[b * 512 + s] = ek;
    skb[b * 512 + s] = ek - sD;
    if (s == 0) {
        out_alpha[b] = alpha;
        alw[b] = alpha;
        float tl = fminf(fmaxf(roundf(alpha * totalD), 1.f), 1000.f);
        tlen[b] = (int)tl;
    }
}

// ---------------------------------------------------------------------------
// Cv = LN(relu(conv1d(alpha*V, cw_w) + cw_b)) over 8 channels
// ---------------------------------------------------------------------------
__global__ __launch_bounds__(256)
void k_cv(const float* __restrict__ V, const float* __restrict__ cww,
          const float* __restrict__ cwb, const float* __restrict__ cwg,
          const float* __restrict__ cwbb, const float* __restrict__ alpha,
          float* __restrict__ cv)
{
    int i = blockIdx.x * 256 + threadIdx.x;
    if (i >= 8192) return;
    int b = i >> 9, s = i & 511;
    float acc[8] = {};
    const float* vb = V + (long long)(b << 9) * 256;
    for (int k = 0; k < 256; ++k) {
        float xm = (s > 0)   ? vb[(s - 1) * 256 + k] : 0.f;
        float x0 = vb[s * 256 + k];
        float xp = (s < 511) ? vb[(s + 1) * 256 + k] : 0.f;
#pragma unroll
        for (int c = 0; c < 8; ++c) {
            const float* w = cww + (c * 256 + k) * 3;
            acc[c] += w[0] * xm + w[1] * x0 + w[2] * xp;
        }
    }
    float al = alpha[b];
    float mean = 0.f;
#pragma unroll
    for (int c = 0; c < 8; ++c) {
        acc[c] = fmaxf(acc[c] * al + cwb[c], 0.f);
        mean += acc[c];
    }
    mean *= 0.125f;
    float var = 0.f;
#pragma unroll
    for (int c = 0; c < 8; ++c) { float d = acc[c] - mean; var += d * d; }
    var *= 0.125f;
    float inv = rsqrtf(var + 1e-5f);
#pragma unroll
    for (int c = 0; c < 8; ++c)
        cv[(long long)i * 8 + c] = (acc[c] - mean) * inv * cwg[c] + cwbb[c];
}

// ---------------------------------------------------------------------------
// Fused 2x conv2d(3x3) + SiLU + 16->1 projection -> attention logits
// Block: (schunk 0..3, t 0..255, b 0..15), 256 threads = 8 waves.
// conv-as-GEMM: K = 9 taps * 16 padded channels = 144 (5 chunks of 32), N = 16.
// im2col A gather: per fragment, two group-uniform contiguous v8h LDS loads.
// ---------------------------------------------------------------------------
DEV v8h conv_gather8(const h16* lds, int kstart, int rbase, int col,
                     int rowStrideHalfs) {
    v8h g = {};
    if (kstart < 144) {
        int tap = kstart >> 4, c0 = kstart & 15;
        int dt = tap / 3, ds = tap - dt * 3;
        g = *(const v8h*)&lds[((rbase + dt) * rowStrideHalfs + (col + ds)) * 16 + c0];
    }
    return g;
}

__global__ __launch_bounds__(256)
void k_mkb(const float* __restrict__ skb, const float* __restrict__ ekb,
           const float* __restrict__ cv, const unsigned char* __restrict__ mask,
           const int* __restrict__ tlen,
           const h16* __restrict__ W1p, const h16* __restrict__ W2p,
           const float* __restrict__ b1, const float* __restrict__ b2,
           const float* __restrict__ lww, const float* __restrict__ lwb,
           float* __restrict__ Wlog)
{
    const int schunk = blockIdx.x;
    const int t = blockIdx.y;
    const int b = blockIdx.z;
    const int s0 = schunk * 128;
    const int tid = threadIdx.x;
    const int wave = tid >> 5;
    const int lane = tid & 31;

    __shared__ __align__(16) h16 fl[5 * 148 * 16];  // feat t-2..t+2, s0-10..s0+137
    __shared__ __align__(16) h16 hl[3 * 144 * 16];  // silu(conv1) t-1..t+1
    __shared__ __align__(16) h16 w1t[16 * 160];     // W1 [n][k]
    __shared__ __align__(16) h16 w2t[16 * 160];
    __shared__ float hsw[128 * 16];                 // silu(conv2) for row t

    const int TL = tlen[b];

    // ---- stage conv weights (already N-major in global): contiguous copy ----
    {
        const v8h* s1 = (const v8h*)W1p;
        const v8h* s2 = (const v8h*)W2p;
        v8h* d1 = (v8h*)w1t;
        v8h* d2 = (v8h*)w2t;
        for (int i = tid; i < 320; i += 256) { d1[i] = s1[i]; d2[i] = s2[i]; }
    }

    // ---- stage feat into LDS (f16, 16-channel padded) ----
    for (int pos = tid; pos < 5 * 148; pos += 256) {
        int tr = pos / 148, si = pos - tr * 148;
        int tg = t - 2 + tr;
        int sg = s0 - 10 + si;
        v8h lo = {}, hi = {};
        if (tg >= 0 && tg < 256 && sg >= 0 && sg < 512) {
            bool am = (tg >= TL) && (mask[b * 512 + sg] != 0);
            float tv = (float)(tg + 1);
            lo[0] = (h16)(am ? 0.f : (tv - skb[b * 512 + sg]));
            lo[1] = (h16)(am ? 0.f : (ekb[b * 512 + sg] - tv));
            const float* cp = &cv[((long long)b * 512 + sg) * 8];
#pragma unroll
            for (int c = 0; c < 6; ++c) lo[2 + c] = (h16)cp[c];
            hi[0] = (h16)cp[6];
            hi[1] = (h16)cp[7];
        }
        *(v8h*)&fl[pos * 16] = lo;
        *(v8h*)&fl[pos * 16 + 8] = hi;
    }
    __syncthreads();

    const int m   = lane & 15;
    const int kb8 = (lane & 16) ? 8 : 0;
    const int mb  = (lane & 16) ? 8 : 0;
    const int n   = lane & 15;
    const float bias1 = b1[n];          // hoisted, loop-invariant per lane
    const float bias2 = b2[n];

    // ---- h1 = silu(conv1(feat)): 3 rows x 9 s-tiles = 27 tile jobs ----
    for (int j = wave; j < 27; j += 8) {
        int r = j / 9, ti = j - r * 9;
        int col = 16 * ti + m + 1;      // fl column for ds=0
        v8f acc = {};
#pragma unroll
        for (int kc = 0; kc < 5; ++kc) {
            int kbase = kc * 32 + kb8;
            v8h g0 = conv_gather8(fl, kbase,      r, col, 148);
            v8h g1 = conv_gather8(fl, kbase + 16, r, col, 148);
            acc = wmma16(cat8(g0, g1), frag_bt160(w1t, kc), acc);
        }
        int tg = t - 1 + r;
#pragma unroll
        for (int i = 0; i < 8; ++i) {
            int sl = 16 * ti + mb + i;
            int sg = s0 - 8 + sl;
            float v = 0.f;              // conv2 zero-padding outside the domain
            if (tg >= 0 && tg < 256 && sg >= 0 && sg < 512)
                v = siluf_(acc[i] + bias1);
            hl[(r * 144 + sl) * 16 + n] = (h16)v;
        }
    }
    __syncthreads();

    // ---- h2 = silu(conv2(h1)): one 16x16 tile per wave ----
    {
        int col = 16 * wave + m + 7;    // hl column for ds=0
        v8f acc = {};
#pragma unroll
        for (int kc = 0; kc < 5; ++kc) {
            int kbase = kc * 32 + kb8;
            v8h g0 = conv_gather8(hl, kbase,      0, col, 144);
            v8h g1 = conv_gather8(hl, kbase + 16, 0, col, 144);
            acc = wmma16(cat8(g0, g1), frag_bt160(w2t, kc), acc);
        }
#pragma unroll
        for (int i = 0; i < 8; ++i)
            hsw[(16 * wave + mb + i) * 16 + n] = siluf_(acc[i] + bias2);
    }
    __syncthreads();

    // ---- 16 -> 1 projection + src mask -> logits ----
    if (tid < 128) {
        int sg = s0 + tid;
        float a = lwb[0];
#pragma unroll
        for (int c = 0; c < 16; ++c) a += hsw[tid * 16 + c] * lww[c];
        bool sm = mask[b * 512 + sg] != 0;
        Wlog[((long long)(b * 256 + t)) * 512 + sg] = sm ? -__builtin_inff() : a;
    }
}

__global__ __launch_bounds__(256)
void k_softmax512(float* __restrict__ W) {
    int row = blockIdx.x;
    int tid = threadIdx.x;
    __shared__ float red[256];
    float* p = W + (long long)row * 512;
    float a = p[tid], bv = p[tid + 256];
    red[tid] = fmaxf(a, bv); __syncthreads();
    for (int st = 128; st > 0; st >>= 1) { if (tid < st) red[tid] = fmaxf(red[tid], red[tid + st]); __syncthreads(); }
    float mx = red[0]; __syncthreads();
    float e0 = (a  == -__builtin_inff()) ? 0.f : __expf(a  - mx);
    float e1 = (bv == -__builtin_inff()) ? 0.f : __expf(bv - mx);
    red[tid] = e0 + e1; __syncthreads();
    for (int st = 128; st > 0; st >>= 1) { if (tid < st) red[tid] += red[tid + st]; __syncthreads(); }
    float s = red[0];
    float inv = (s > 0.f) ? fastrcp(s) : 0.f;
    p[tid] = e0 * inv; p[tid + 256] = e1 * inv;
}

__global__ void k_textmask(const int* __restrict__ tlen, float* __restrict__ out) {
    int i = blockIdx.x * blockDim.x + threadIdx.x;
    if (i < 16 * 256) out[i] = ((i & 255) >= tlen[i >> 8]) ? 1.f : 0.f;
}

// ---------------------------------------------------------------------------
// Host orchestration
// ---------------------------------------------------------------------------
extern "C" void kernel_launch(void* const* d_in, const int* in_sizes, int n_in,
                              void* d_out, int out_size, void* d_ws, size_t ws_size,
                              hipStream_t stream)
{
    const float* content = (const float*)d_in[0];
    const float* style   = (const float*)d_in[1];
    const unsigned char* mask = (const unsigned char*)d_in[2];
    const float* rhythm  = (const float*)d_in[3];
    const float* mel     = (const float*)d_in[4];
    const float* in_w    = (const float*)d_in[5];
    const float* in_b    = (const float*)d_in[6];
    const float *glu_w[3], *glu_b[3], *lck[3], *out_w[3], *out_b[3], *ln_g[3], *ln_b[3];
    for (int L = 0; L < 3; ++L) {
        int base = 7 + L * 7;
        glu_w[L] = (const float*)d_in[base + 0];
        glu_b[L] = (const float*)d_in[base + 1];
        lck[L]   = (const float*)d_in[base + 2];
        out_w[L] = (const float*)d_in[base + 3];
        out_b[L] = (const float*)d_in[base + 4];
        ln_g[L]  = (const float*)d_in[base + 5];
        ln_b[L]  = (const float*)d_in[base + 6];
    }
    const float* p1_w   = (const float*)d_in[28];
    const float* p1_b   = (const float*)d_in[29];
    const float* p2_w   = (const float*)d_in[30];
    const float* p2_b   = (const float*)d_in[31];
    const float* cw_w   = (const float*)d_in[32];
    const float* cw_b   = (const float*)d_in[33];
    const float* cw_g   = (const float*)d_in[34];
    const float* cw_bb  = (const float*)d_in[35];
    const float* mkb_w1 = (const float*)d_in[36];
    const float* mkb_b1 = (const float*)d_in[37];
    const float* mkb_w2 = (const float*)d_in[38];
    const float* mkb_b2 = (const float*)d_in[39];
    const float* lw_w   = (const float*)d_in[40];
    const float* lw_b   = (const float*)d_in[41];
    const float* lnf_g  = (const float*)d_in[42];
    const float* lnf_b  = (const float*)d_in[43];

    float* out    = (float*)d_out;
    float* out_ce = out;                 // 16*256*256
    float* out_tm = out + 1048576;       // 16*256
    float* out_W  = out + 1052672;       // 16*256*512
    float* out_al = out + 3149824;       // 16
    float* out_sD = out + 3149840;       // 16*512

    float* ws   = (float*)d_ws;
    float* xbuf = ws;                    // 8192*256
    float* hglu = xbuf + 2097152;        // 8192*512
    float* hc   = hglu + 4194304;        // 8192*256
    float* cer  = hc   + 2097152;        // 16*256*256
    float* cvb  = cer  + 1048576;        // 16*512*8
    float* skb  = cvb  + 65536;          // 16*512
    float* ekb  = skb  + 8192;           // 16*512
    float* alw  = ekb  + 8192;           // 16
    int*   tlen = (int*)(alw + 16);      // 16
    h16*   fw   = (h16*)(tlen + 16);
    h16* in_wh  = fw;                    // 384*256   [n*384+k]
    h16* glu_h0 = in_wh + 98304;         // 3 * 131072 [n*256+k]
    h16* out_h0 = glu_h0 + 3 * 131072;   // 3 * 65536  [n*256+k]
    h16* p2_h   = out_h0 + 3 * 65536;    // 196608     [n*768+k]
    h16* w1p    = p2_h + 196608;         // 2560       [n*160+k]
    h16* w2p    = w1p + 2560;            // 2560
    h16* sPh    = w2p + 2560;            // 16 * (256 n * 512 s) transposed

    float* hdw = hglu;                   // reuse after GLU gate
    float* tmp = hglu + 2097152;

    // --- weight conversion / repack (N-major f16) ---
    k_cvt_t<<<(98304 + 255) / 256, 256, 0, stream>>>(in_w, in_wh, 384, 256);
    for (int L = 0; L < 3; ++L) {
        k_cvt_t<<<(131072 + 255) / 256, 256, 0, stream>>>(glu_w[L], glu_h0 + L * 131072, 256, 512);
        k_cvt_t<<<(65536 + 255) / 256, 256, 0, stream>>>(out_w[L], out_h0 + L * 65536, 256, 256);
    }
    k_pack_p2<<<(256 * 768 + 255) / 256, 256, 0, stream>>>(p2_w, p2_h);
    k_pack_mkb<<<10, 256, 0, stream>>>(mkb_w1, w1p, 10);
    k_pack_mkb<<<10, 256, 0, stream>>>(mkb_w2, w2p, 16);

    // --- input linear: x = [content ; style] @ in_w + in_b ---
    gemm_wmma<1, true, false, true, false><<<dim3(4, 128, 1), 256, 0, stream>>>(
        content, style, in_wh, in_b, xbuf, nullptr, nullptr,
        8192, 256, 384, 0, 0, 0);

    // --- 3 lightweight-conv blocks ---
    for (int L = 0; L < 3; ++L) {
        gemm_wmma<0, true, false, true, false><<<dim3(8, 128, 1), 256, 0, stream>>>(
            xbuf, nullptr, glu_h0 + L * 131072, glu_b[L], hglu, nullptr, nullptr,
            8192, 512, 256, 0, 0, 0);
        k_glu_mask<<<8192, 256, 0, stream>>>(hglu, mask, hc);
        k_dwconv<<<8192, 256, 0, stream>>>(hc, lck[L], hdw);
        gemm_wmma<0, true, false, true, false><<<dim3(4, 128, 1), 256, 0, stream>>>(
            hdw, nullptr, out_h0 + L * 65536, out_b[L], tmp, nullptr, nullptr,
            8192, 256, 256, 0, 0, 0);
        k_add_ln_mask<<<8192, 256, 0, stream>>>(xbuf, tmp, ln_g[L], ln_b[L], mask, xbuf);
    }

    // --- duration head: D, alpha, sD, cumsum, text_len ---
    k_duration<<<16, 512, 0, stream>>>(xbuf, p1_w, p1_b, mask, rhythm, mel,
                                       out_al, out_sD, skb, ekb, tlen, alw);

    // --- Cv (8-channel conv + relu + LN) ---
    k_cv<<<32, 256, 0, stream>>>(xbuf, cw_w, cw_b, cw_g, cw_bb, alw, cvb);

    // --- sP = alpha*(conv1d(V,p2_w)+b) via im2col WMMA GEMM, f16 transposed ---
    gemm_wmma<2, true, true, false, true><<<dim3(4, 128, 1), 256, 0, stream>>>(
        xbuf, nullptr, p2_h, p2_b, nullptr, sPh, alw,
        8192, 256, 768, 0, 0, 0);

    // --- fused conv2d x2 + projection -> logits ---
    k_mkb<<<dim3(4, 256, 16), 256, 0, stream>>>(
        skb, ekb, cvb, mask, tlen, w1p, w2p, mkb_b1, mkb_b2, lw_w, lw_b, out_W);

    // --- row softmax over S=512 (in place on out_W) ---
    k_softmax512<<<4096, 256, 0, stream>>>(out_W);

    // --- ce = W @ sP per batch (batched WMMA GEMM, async-staged B) ---
    gemm_wmma<0, false, false, true, false><<<dim3(4, 4, 16), 256, 0, stream>>>(
        out_W, nullptr, sPh, nullptr, cer, nullptr, nullptr,
        256, 256, 512, 131072LL, 131072LL, 65536LL);

    // --- final LN + text mask ---
    k_ce_ln<<<4096, 256, 0, stream>>>(cer, lnf_g, lnf_b, tlen, out_ce);
    k_textmask<<<16, 256, 0, stream>>>(tlen, out_tm);
}
